// AAConv_78108275245489
// MI455X (gfx1250) — matrix operand order repaired and checked by reading, not compiled, not run on hardware
//
#include <hip/hip_runtime.h>

// ---------------------------------------------------------------------------
// AAConv (attention-augmented conv) for gfx1250 / MI455X.
// B=8, Cin=64, H=W=32, Cout=128, DK=DV=64, NH=8, DKH=DVH=8, CCONV=64.
//
// Flash-style attention with v_wmma_f32_16x16x32_f16:
//  - height rel-logits folded into the K tile (rank-1, shared A factor)
//  - width rel-logits via transposed LDS table, one ds_load_b128 per tile
//  - softmax denominator accumulated as a ones-column of V inside the PV WMMA
//  - no cross-lane reductions in the hot loop (shift-free exp is safe here)
// ---------------------------------------------------------------------------

typedef __attribute__((ext_vector_type(16))) _Float16 v16h;
typedef __attribute__((ext_vector_type(8)))  _Float16 h8;
typedef __attribute__((ext_vector_type(8)))  float    v8f;

#define BB   8
#define CIN  64
#define HW   1024
#define NH_  8
#define DKH_ 8
#define CCONV 64

// ---------------------------------------------------------------------------
// Kernel 1: 1x1 QKV conv.  q is pre-scaled by DKH^-0.5.
//   q16, k16: [((b*NH + n)*HW + pos)*8 + d]          (row per position)
//   v16T    : [(b*NH + n)*8*HW + d*HW + pos]         (transposed: row per d)
// ---------------------------------------------------------------------------
__global__ __launch_bounds__(256) void qkv_prep(
    const float* __restrict__ x, const float* __restrict__ qkv_w,
    const float* __restrict__ qkv_b,
    _Float16* __restrict__ q16, _Float16* __restrict__ k16,
    _Float16* __restrict__ v16T)
{
  int idx = blockIdx.x * 256 + threadIdx.x;          // over B*HW*192
  if (idx >= BB * HW * 192) return;
  int c   = idx % 192;
  int pos = (idx / 192) % HW;
  int b   = idx / (192 * HW);

  const float* xp = x + (b * CIN) * HW + pos;        // x[b, ci, pos], stride HW
  const float* wr = qkv_w + c * CIN;                 // (192, 64, 1, 1)
  float acc = qkv_b[c];
  #pragma unroll 8
  for (int ci = 0; ci < CIN; ++ci) acc += wr[ci] * xp[ci * HW];

  if (c < 64) {                                      // q, scaled
    acc *= 0.35355339059327373f;                     // 8^-0.5
    int n = c >> 3, d = c & 7;
    q16[((b * NH_ + n) * HW + pos) * 8 + d] = (_Float16)acc;
  } else if (c < 128) {                              // k
    int c2 = c - 64; int n = c2 >> 3, d = c2 & 7;
    k16[((b * NH_ + n) * HW + pos) * 8 + d] = (_Float16)acc;
  } else {                                           // v (transposed layout)
    int c2 = c - 128; int n = c2 >> 3, d = c2 & 7;
    v16T[(size_t)(b * NH_ + n) * 8 * HW + d * HW + pos] = (_Float16)acc;
  }
}

// ---------------------------------------------------------------------------
// Kernel 2: attention, one wave (32 lanes) per 16-query tile.
// Grid: B*NH*64 waves = 4096 -> 512 blocks of 8 waves.
// Writes attn_pre[((b*NH+n)*HW + q)*8 + d]  (f32, softmax-normalized).
// ---------------------------------------------------------------------------
__global__ __launch_bounds__(256) void attn_kernel(
    const _Float16* __restrict__ q16, const _Float16* __restrict__ k16,
    const _Float16* __restrict__ v16T,
    const float* __restrict__ key_rel_w, const float* __restrict__ key_rel_h,
    float* __restrict__ attn_pre)
{
  __shared__ __align__(16) _Float16 s_relwT[8][32][16]; // per-wave RW[wk][q]
  __shared__ __align__(16) _Float16 s_p[8][16][16];     // per-wave P staging

  const int wv   = threadIdx.x >> 5;
  const int lane = threadIdx.x & 31;
  const int gw   = blockIdx.x * 8 + wv;    // global wave id, 0..4095
  const int b    = gw >> 9;                // / (NH*64)
  const int n    = (gw >> 6) & 7;
  const int t    = gw & 63;                // query tile
  const int q0   = t * 16;
  const int wqb  = (t & 1) * 16;           // wq base for this tile (16 cols/tile)
  const int hq   = t >> 1;                 // constant row for this tile

  const _Float16* qb  = q16  + (size_t)(b * NH_ + n) * HW * 8;
  const _Float16* kb  = k16  + (size_t)(b * NH_ + n) * HW * 8;
  const _Float16* vbT = v16T + (size_t)(b * NH_ + n) * 8 * HW;

  // --- width rel-logit table, transposed: RW[wk][i] (q already scaled) ---
  for (int e = lane; e < 512; e += 32) {
    int wk = e >> 4, i = e & 15;
    const _Float16* qr = qb + (q0 + i) * 8;
    const float* rw = key_rel_w + (wk - (wqb + i) + 31) * DKH_; // in [0,62]
    float aw = 0.f;
    #pragma unroll
    for (int d = 0; d < 8; ++d) aw += (float)qr[d] * rw[d];
    s_relwT[wv][wk][i] = (_Float16)aw;
  }
  asm volatile("s_wait_dscnt 0" ::: "memory");   // wave-private LDS RAW

  const int  M   = lane & 15;
  const bool hi  = lane >= 16;                   // upper 16 lanes
  const int  N   = lane & 15;
  const _Float16 Z   = (_Float16)0.0f;
  const _Float16 ONE = (_Float16)1.0f;

  // --- Q tile in A layout: lanes 0-15 carry K=0..7 at slots 0..7, rest 0 ---
  const h8 qrow = *(const h8*)(qb + (q0 + M) * 8);   // one b128 load
  v16h a_q;
  #pragma unroll
  for (int h = 0; h < 8; ++h) a_q[h] = hi ? Z : qrow[h];
  #pragma unroll
  for (int h = 8; h < 16; ++h) a_q[h] = Z;

  // O accumulator; column 8 accumulates the softmax denominator (ones-col).
  v8f o = {0.f, 0.f, 0.f, 0.f, 0.f, 0.f, 0.f, 0.f};

  const bool vz   = hi | (N >= 8);               // V pad columns / rows
  const _Float16 vfill = (!hi && N == 8) ? ONE : Z;   // ones-column for l

  for (int hk = 0; hk < 32; ++hk) {              // key row (16x2 keys each)
    // wave-uniform height rel row: fold into every key row of B
    const float* krh = key_rel_h + (hk - hq + 31) * DKH_;   // in [0,62]
    _Float16 krh16[8];
    #pragma unroll
    for (int d = 0; d < 8; ++d) krh16[d] = (_Float16)krh[d];

    #pragma unroll
    for (int jj = 0; jj < 2; ++jj) {
      const int j = hk * 2 + jj;                 // key tile 0..63
      if (j < 63) {
        __builtin_prefetch(kb + (j + 1) * 16 * 8, 0, 1);
        __builtin_prefetch(vbT + (N & 7) * HW + (j + 1) * 16, 0, 1);
      }
      // --- B = K^T tile + folded RH: lanes 0-15 carry key-row N halves ---
      const h8 krow = *(const h8*)(kb + (j * 16 + N) * 8);  // one b128 load
      v16h b_k;
      #pragma unroll
      for (int h = 0; h < 8; ++h) b_k[h] = hi ? Z : (_Float16)(krow[h] + krh16[h]);
      #pragma unroll
      for (int h = 8; h < 16; ++h) b_k[h] = Z;

      v8f s = {0.f, 0.f, 0.f, 0.f, 0.f, 0.f, 0.f, 0.f};
      s = __builtin_amdgcn_wmma_f32_16x16x32_f16(false, a_q, false, b_k,
                                                 (short)0, s, false, false);

      // --- width rel for this lane's 8 rows: one ds_load_b128 ---
      const int wk = jj * 16 + N;                // == kidx & 31
      const h8 rwv = *(const h8*)(&s_relwT[wv][wk][hi ? 8 : 0]);

      // --- exp (shift-free; logits are O(1)) and stage P rows ---
      #pragma unroll
      for (int r = 0; r < 8; ++r) {
        float p = __expf(s[r] + (float)rwv[r]);
        s_p[wv][r + (hi ? 8 : 0)][N] = (_Float16)p;
      }
      asm volatile("s_wait_dscnt 0" ::: "memory");
      // lanes 0-15 need P[M][0..15] at slots 0..15, lanes 16-31 zero: 2x b128 LDS load
      const h8 prow = *(const h8*)(&s_p[wv][M][0]);
      const h8 prow1 = *(const h8*)(&s_p[wv][M][8]);
      v16h a_p;
      #pragma unroll
      for (int h = 0; h < 8; ++h) a_p[h] = hi ? Z : prow[h];
      #pragma unroll
      for (int h = 8; h < 16; ++h) a_p[h] = hi ? Z : prow1[h - 8];

      // --- B = V tile (+ ones-column at N=8 for the denominator) ---
      const _Float16* vrow = vbT + (N & 7) * HW + j * 16;   // 32B contiguous
      const h8 v0 = *(const h8*)(vrow);
      const h8 v1 = *(const h8*)(vrow + 8);
      v16h b_v;
      #pragma unroll
      for (int h = 0; h < 8; ++h)  b_v[h]     = vz ? vfill : v0[h];
      #pragma unroll
      for (int h = 0; h < 8; ++h)  b_v[h + 8] = vz ? vfill : v1[h];

      o = __builtin_amdgcn_wmma_f32_16x16x32_f16(false, a_p, false, b_v,
                                                 (short)0, o, false, false);
    }
  }

  // --- epilogue: broadcast denominator from column 8, normalize, store ---
  const int srcl = (lane & 16) | 8;              // lane holding column 8
  #pragma unroll
  for (int r = 0; r < 8; ++r) {
    float l = __shfl(o[r], srcl, 32);
    float val = o[r] / l;
    if (N < 8) {
      int i = r + (hi ? 8 : 0);
      attn_pre[((size_t)(b * NH_ + n) * HW + q0 + i) * 8 + N] = val;
    }
  }
}

// ---------------------------------------------------------------------------
// Kernel 3: 3x3 conv branch (channels 0..63) + 1x1 attn projection over the
// torch-.view channels (64..127), concatenated into d_out (B,128,32,32).
// ---------------------------------------------------------------------------
__global__ __launch_bounds__(256) void out_kernel(
    const float* __restrict__ x, const float* __restrict__ conv_w,
    const float* __restrict__ conv_b, const float* __restrict__ attn_w,
    const float* __restrict__ attn_b, const float* __restrict__ attn_pre,
    float* __restrict__ out)
{
  int idx = blockIdx.x * 256 + threadIdx.x;          // over B*128*HW
  if (idx >= BB * 128 * HW) return;
  int pos = idx & 1023;
  int c   = (idx >> 10) & 127;
  int b   = idx >> 17;
  int h = pos >> 5, w = pos & 31;

  float acc;
  if (c < CCONV) {                                   // 3x3 conv, pad 1
    acc = conv_b[c];
    const float* wp = conv_w + c * CIN * 9;
    for (int ci = 0; ci < CIN; ++ci) {
      const float* xp = x + (size_t)(b * CIN + ci) * HW;
      const float* wc = wp + ci * 9;
      #pragma unroll
      for (int kh = 0; kh < 3; ++kh) {
        int hh = h + kh - 1;
        if ((unsigned)hh >= 32u) continue;
        #pragma unroll
        for (int kw = 0; kw < 3; ++kw) {
          int ww = w + kw - 1;
          if ((unsigned)ww >= 32u) continue;
          acc += wc[kh * 3 + kw] * xp[hh * 32 + ww];
        }
      }
    }
  } else {                                           // 1x1 attn projection
    int c2 = c - CCONV;
    acc = attn_b[c2];
    const float* ap = attn_pre + (size_t)b * 64 * HW;   // viewed (64, 1024)
    const float* wp = attn_w + c2 * 64;
    #pragma unroll 8
    for (int ci = 0; ci < 64; ++ci) acc += wp[ci] * ap[ci * HW + pos];
  }
  out[idx] = acc;
}

// ---------------------------------------------------------------------------
extern "C" void kernel_launch(void* const* d_in, const int* in_sizes, int n_in,
                              void* d_out, int out_size, void* d_ws, size_t ws_size,
                              hipStream_t stream)
{
  const float* x         = (const float*)d_in[0];
  const float* conv_w    = (const float*)d_in[1];
  const float* conv_b    = (const float*)d_in[2];
  const float* qkv_w     = (const float*)d_in[3];
  const float* qkv_b     = (const float*)d_in[4];
  const float* attn_w    = (const float*)d_in[5];
  const float* attn_b    = (const float*)d_in[6];
  const float* key_rel_w = (const float*)d_in[7];
  const float* key_rel_h = (const float*)d_in[8];

  // workspace: q16 | k16 | v16T (f16, 1 MB each) | attn_pre (f32, 2 MB)
  const size_t QKV_ELEMS = (size_t)BB * NH_ * HW * 8;   // 524288
  _Float16* q16  = (_Float16*)d_ws;
  _Float16* k16  = q16 + QKV_ELEMS;
  _Float16* v16T = k16 + QKV_ELEMS;
  float* attn_pre = (float*)(v16T + QKV_ELEMS);
  float* out = (float*)d_out;

  {
    int total = BB * HW * 192;
    qkv_prep<<<(total + 255) / 256, 256, 0, stream>>>(x, qkv_w, qkv_b,
                                                      q16, k16, v16T);
  }
  {
    // B*NH*64 query tiles = 4096 waves; 8 waves per block
    attn_kernel<<<512, 256, 0, stream>>>(q16, k16, v16T, key_rel_w, key_rel_h,
                                         attn_pre);
  }
  {
    int total = BB * 128 * HW;
    out_kernel<<<(total + 255) / 256, 256, 0, stream>>>(x, conv_w, conv_b,
                                                        attn_w, attn_b,
                                                        attn_pre, out);
  }
}
